// SlicedUnmix_57732950392841
// MI455X (gfx1250) — matrix operand order repaired and checked by reading, not compiled, not run on hardware
//
#include <hip/hip_runtime.h>
#include <cstdint>
#include <cstddef>

typedef __bf16 bf16;
typedef __bf16 v8bf  __attribute__((ext_vector_type(8)));
typedef __bf16 v16bf __attribute__((ext_vector_type(16)));
typedef float  v8f   __attribute__((ext_vector_type(8)));

#define NB  16
#define NCH 2
#define NF  32
#define NS  64
#define NT  112
#define HOP 56
#define TOA 3640   /* (NS+1)*HOP */

// ---------------------------------------------------------------------------
// Preprocess: overlap_add + (x+mean)*scale, complex f32 -> channels-last bf16
// X0 layout: [b][f][u][c], CinPad=16 (c0=re ch0, c1=im ch0, c2=re ch1, c3=im ch1)
// ---------------------------------------------------------------------------
__global__ __launch_bounds__(256) void k_preprocess(
    const float* __restrict__ x, const float* __restrict__ mean,
    const float* __restrict__ scale, bf16* __restrict__ X0)
{
  long idx = (long)blockIdx.x * blockDim.x + threadIdx.x;
  const long total = (long)NB * NF * TOA;
  if (idx >= total) return;
  int u = (int)(idx % TOA); long r = idx / TOA;
  int f = (int)(r % NF);    int b = (int)(r / NF);
  int s = u / HOP, h = u - s * HOP;

  v16bf out;
  #pragma unroll
  for (int e = 0; e < 16; ++e) out[e] = (bf16)0.f;

  #pragma unroll
  for (int c = 0; c < NCH; ++c) {
    float re = 0.f, im = 0.f;
    if (s < NS) {
      long i0 = ((((long)b * NCH + c) * NF + f) * NS + s) * NT + h;
      re += x[2 * i0]; im += x[2 * i0 + 1];
    }
    if (s >= 1) {
      long i1 = ((((long)b * NCH + c) * NF + f) * NS + (s - 1)) * NT + (HOP + h);
      re += x[2 * i1]; im += x[2 * i1 + 1];
    }
    re = (re + mean[f]) * scale[f];
    im = im * scale[f];
    out[2 * c]     = (bf16)re;
    out[2 * c + 1] = (bf16)im;
  }
  *(v16bf*)(X0 + idx * 16) = out;
}

// ---------------------------------------------------------------------------
// Weight expansion: complex w (CoutR,CinR,KH,KW) [or convT: (CinR,CoutR,KH,KW),
// flipped] -> real A-matrix bf16 [Mpad][Kpad], K = tap-major, channel-minor.
// Row 2co = [wR | -wI], row 2co+1 = [wI | wR]. Also emits f32 bias[Mpad].
// ---------------------------------------------------------------------------
__global__ __launch_bounds__(256) void k_expand(
    const float* __restrict__ w, const float* __restrict__ bias,
    bf16* __restrict__ Wm, float* __restrict__ bv,
    int CoutR, int CinR, int KH, int KW, int transposeFlip,
    int Mpad, int CinPad, int tapsPad)
{
  long Kpad = (long)tapsPad * CinPad;
  long idx = (long)blockIdx.x * blockDim.x + threadIdx.x;
  if (idx >= (long)Mpad * Kpad) return;
  int m = (int)(idx / Kpad);
  int k = (int)(idx - (long)m * Kpad);
  int tap = k / CinPad, c = k - tap * CinPad;
  float val = 0.f;
  if (tap < KH * KW && m < 2 * CoutR && c < 2 * CinR) {
    int kh = tap / KW, kw = tap - (tap / KW) * KW;
    int co = m >> 1, mi = m & 1, ci = c >> 1, cim = c & 1;
    long sidx;
    if (!transposeFlip)
      sidx = (((long)co * CinR + ci) * KH + kh) * KW + kw;
    else
      sidx = (((long)ci * CoutR + co) * KH + (KH - 1 - kh)) * KW + (KW - 1 - kw);
    float re = w[2 * sidx], im = w[2 * sidx + 1];
    val = mi ? (cim ? re : im) : (cim ? -im : re);
  }
  Wm[idx] = (bf16)val;
  if (k == 0) {
    float bval = 0.f;
    if (m < 2 * CoutR) bval = (m & 1) ? bias[2 * (m >> 1) + 1] : bias[2 * (m >> 1)];
    bv[m] = bval;
  }
}

// ---------------------------------------------------------------------------
// Implicit-GEMM conv/convT via V_WMMA_F32_16X16X32_BF16.
// Grid: x = time-tile groups (8 waves/block), y = Fout row, z = batch.
// One wave = one 16-wide time tile x ALL MT output-channel tiles.
// Weights staged in 256-wide K slabs into LDS (shared by all 8 waves),
// A-fragments then come from ds_load; activations stream from global.
// ---------------------------------------------------------------------------
struct ConvGeo {
  const bf16* X; const bf16* W; const float* bias; bf16* Y;
  int B, Fin, Tin, CinPad;
  int Fout, Tout, CoutPad, Cstore;
  int KW, dil_t, f_off, t_off;
  int ntaps, tapsPad, Kpad;
  int applyTanh;
};

#define KSUP 256   /* K-slab width staged in LDS (multiple of 32) */

template <int MT>
__global__ __launch_bounds__(256) void k_conv_wmma(ConvGeo g)
{
  extern __shared__ char smem_raw[];
  bf16* smem = (bf16*)smem_raw;                 // [MT*16][KSUP]

  int wid  = threadIdx.x >> 5;
  int lane = threadIdx.x & 31;
  int hh = lane >> 4, ln = lane & 15;
  int nt = blockIdx.x * 8 + wid;                // may exceed Ntiles (tail block):
  int f  = blockIdx.y;                          // then all loads/stores predicate off
  int b  = blockIdx.z;
  int t  = nt * 16 + ln;                        // output time column for this lane

  v8f acc[MT];
  #pragma unroll
  for (int mt = 0; mt < MT; ++mt) {
    #pragma unroll
    for (int e = 0; e < 8; ++e) acc[mt][e] = 0.f;
  }

  const int Mrows = MT * 16;
  int tap = 0, kh = 0, kw = 0, c = 0;           // incremental K cursor (no divides)

  for (int ks = 0; ks < g.Kpad; ks += KSUP) {
    int ksz = g.Kpad - ks; if (ksz > KSUP) ksz = KSUP;

    __syncthreads();                            // previous slab fully consumed
    for (int row = wid; row < Mrows; row += 8) {
      if (lane * 8 < ksz) {
        v8bf v = *(const v8bf*)(g.W + (long)row * g.Kpad + ks + lane * 8);
        *(v8bf*)(smem + row * KSUP + lane * 8) = v;
      }
    }
    __syncthreads();                            // slab visible to all waves

    int nch = ksz >> 4;
    for (int s2 = 0; s2 < nch; s2 += 2) {
      int kh0 = kh, kw0 = kw, c0 = c, tap0 = tap;
      c += 16; if (c == g.CinPad) { c = 0; ++tap; if (++kw == g.KW) { kw = 0; ++kh; } }
      int kh1 = kh, kw1 = kw, c1 = c, tap1 = tap;
      c += 16; if (c == g.CinPad) { c = 0; ++tap; if (++kw == g.KW) { kw = 0; ++kh; } }

      // B fragment: lane half hh owns K = base + hh*16 + e -> chunk (s2+hh)
      int khs = hh ? kh1 : kh0;
      int kws = hh ? kw1 : kw0;
      int cs  = hh ? c1  : c0;
      int tps = hh ? tap1 : tap0;
      int fi = f + khs + g.f_off;
      int ti = t + kws * g.dil_t + g.t_off;
      v16bf bfrag;
      #pragma unroll
      for (int e = 0; e < 16; ++e) bfrag[e] = (bf16)0.f;
      if (tps < g.ntaps && (unsigned)fi < (unsigned)g.Fin && (unsigned)ti < (unsigned)g.Tin)
        bfrag = *(const v16bf*)(g.X + (((long)(b * g.Fin + fi)) * g.Tin + ti) * g.CinPad + cs);

      // prefetch next iteration's activation (cursor now holds next chunk-pair)
      {
        int fi2 = f + kh + g.f_off;
        int ti2 = t + kw * g.dil_t + g.t_off;
        if (tap < g.ntaps && (unsigned)fi2 < (unsigned)g.Fin && (unsigned)ti2 < (unsigned)g.Tin)
          __builtin_prefetch(g.X + (((long)(b * g.Fin + fi2)) * g.Tin + ti2) * g.CinPad + c, 0, 1);
      }

      // A fragments from LDS: per lane K = 16*hh + e
      const bf16* wp = smem + ln * KSUP + s2 * 16;
      #pragma unroll
      for (int mt = 0; mt < MT; ++mt) {
        const bf16* wpm = wp + mt * 16 * KSUP;
        v8bf lo = *(const v8bf*)(wpm + hh * 16);
        v8bf hi = *(const v8bf*)(wpm + hh * 16 + 8);
        v16bf afrag = __builtin_shufflevector(lo, hi, 0,1,2,3,4,5,6,7,8,9,10,11,12,13,14,15);
        acc[mt] = __builtin_amdgcn_wmma_f32_16x16x32_bf16(
            false, afrag, false, bfrag, (short)0, acc[mt], false, false);
      }
    }
  }

  // Epilogue: bias + tanh, f32 -> bf16, channels-last store.
  // D layout: lane (hh,ln): VGPR rr -> M = rr + 8*hh, N = ln.
  if (t < g.Tout) {
    bf16* yrow = g.Y + (((long)(b * g.Fout + f)) * g.Tout + t) * g.CoutPad;
    #pragma unroll
    for (int mt = 0; mt < MT; ++mt) {
      int chbase = mt * 16 + hh * 8;
      if (chbase < g.Cstore) {
        if (chbase + 8 <= g.Cstore) {
          v8bf ov;
          #pragma unroll
          for (int rr = 0; rr < 8; ++rr) {
            float v = acc[mt][rr] + g.bias[chbase + rr];
            if (g.applyTanh) v = tanhf(v);
            ov[rr] = (bf16)v;
          }
          *(v8bf*)(yrow + chbase) = ov;
        } else {
          #pragma unroll
          for (int rr = 0; rr < 8; ++rr) {
            if (chbase + rr < g.Cstore) {
              float v = acc[mt][rr] + g.bias[chbase + rr];
              if (g.applyTanh) v = tanhf(v);
              yrow[chbase + rr] = (bf16)v;
            }
          }
        }
      }
    }
  }
}

// ---------------------------------------------------------------------------
// fin convT (kernel 1x3, stride (1,2)) + crop + multiply-by-mix + store f32 c64
// ---------------------------------------------------------------------------
__global__ __launch_bounds__(256) void k_fin(
    const bf16* __restrict__ X4, const float* __restrict__ finw,
    const float* __restrict__ finb, const float* __restrict__ xin,
    float* __restrict__ out, int br)
{
  long idx = (long)blockIdx.x * blockDim.x + threadIdx.x;
  const long total = (long)NB * NCH * NF * NS * NT;
  if (idx >= total) return;
  int tt = (int)(idx % NT); long r = idx / NT;
  int s  = (int)(r % NS);   r /= NS;
  int f  = (int)(r % NF);   r /= NF;
  int co = (int)(r % NCH);  int b = (int)(r / NCH);
  int tout = s * NT + tt;

  float orr = finb[2 * co], oii = finb[2 * co + 1];
  #pragma unroll
  for (int ci = 0; ci < 2; ++ci) {
    #pragma unroll
    for (int kw = 0; kw < 3; ++kw) {
      int p = tout + kw - 2;
      if (p >= 0 && (p & 1) == 0) {
        int j = p >> 1;
        if (j < TOA) {
          long wi2 = ((long)ci * 2 + co) * 3 + (2 - kw);   // w2 = flip+transpose of fin_w
          float wr = finw[2 * wi2], wim = finw[2 * wi2 + 1];
          const bf16* xp = X4 + (((long)(b * NF + f)) * TOA + j) * 4 + 2 * ci;
          float xr = (float)xp[0], xi = (float)xp[1];
          orr += wr * xr - wim * xi;
          oii += wr * xi + wim * xr;
        }
      }
    }
  }
  long midx = ((((long)b * NCH + co) * NF + f) * (long)NS + s) * NT + tt;
  float mr = xin[2 * midx], mi = xin[2 * midx + 1];
  long oidx = (long)br * total + midx;
  out[2 * oidx]     = orr * mr - oii * mi;
  out[2 * oidx + 1] = orr * mi + oii * mr;
}

// ---------------------------------------------------------------------------
static inline int cdiv_i(long a, long b) { return (int)((a + b - 1) / b); }

extern "C" void kernel_launch(void* const* d_in, const int* in_sizes, int n_in,
                              void* d_out, int out_size, void* d_ws, size_t ws_size,
                              hipStream_t stream)
{
  (void)in_sizes; (void)n_in; (void)out_size; (void)ws_size;
  const float* x     = (const float*)d_in[0];
  const float* mean  = (const float*)d_in[41];
  const float* scale = (const float*)d_in[42];

  char* ws = (char*)d_ws;
  size_t off = 0;
  auto sub = [&](size_t bytes) -> char* {
    off = (off + 255) & ~(size_t)255;
    char* p = ws + off; off += bytes; return p;
  };

  // activations (bf16, channels-last). X3 aliases X1 (enc0 out dead after enc1).
  bf16* X0 = (bf16*)sub((size_t)NB * NF * TOA * 16 * 2);   // (16,32,3640,16)
  bf16* X1 = (bf16*)sub((size_t)NB * 28 * 3616 * 64 * 2);  // enc0 out / dec0 out
  bf16* X2 = (bf16*)sub((size_t)NB * 24 * 3592 * 112 * 2); // enc1 out
  bf16* X4 = (bf16*)sub((size_t)NB * 32 * 3640 * 4 * 2);   // dec1 out

  bf16*  Wenc0 = (bf16*)sub((size_t)64 * 1056 * 2);
  bf16*  Wenc1 = (bf16*)sub((size_t)112 * 4160 * 2);
  bf16*  Wdec0 = (bf16*)sub((size_t)64 * 7392 * 2);
  bf16*  Wdec1 = (bf16*)sub((size_t)16 * 4160 * 2);
  float* Benc0 = (float*)sub(64 * 4);
  float* Benc1 = (float*)sub(112 * 4);
  float* Bdec0 = (float*)sub(64 * 4);
  float* Bdec1 = (float*)sub(16 * 4);

  {
    long tot = (long)NB * NF * TOA;
    k_preprocess<<<cdiv_i(tot, 256), 256, 0, stream>>>(x, mean, scale, X0);
  }

  for (int br = 0; br < 4; ++br) {
    const float* enc0_w = (const float*)d_in[1 + br * 10 + 0];
    const float* enc0_b = (const float*)d_in[1 + br * 10 + 1];
    const float* enc1_w = (const float*)d_in[1 + br * 10 + 2];
    const float* enc1_b = (const float*)d_in[1 + br * 10 + 3];
    const float* dec0_w = (const float*)d_in[1 + br * 10 + 4];
    const float* dec0_b = (const float*)d_in[1 + br * 10 + 5];
    const float* dec1_w = (const float*)d_in[1 + br * 10 + 6];
    const float* dec1_b = (const float*)d_in[1 + br * 10 + 7];
    const float* fin_w  = (const float*)d_in[1 + br * 10 + 8];
    const float* fin_b  = (const float*)d_in[1 + br * 10 + 9];

    k_expand<<<cdiv_i((long)64 * 1056, 256), 256, 0, stream>>>(
        enc0_w, enc0_b, Wenc0, Benc0, 25, 2, 5, 13, 0, 64, 16, 66);
    k_expand<<<cdiv_i((long)112 * 4160, 256), 256, 0, stream>>>(
        enc1_w, enc1_b, Wenc1, Benc1, 55, 25, 5, 13, 0, 112, 64, 65);
    k_expand<<<cdiv_i((long)64 * 7392, 256), 256, 0, stream>>>(
        dec0_w, dec0_b, Wdec0, Bdec0, 25, 55, 5, 13, 1, 64, 112, 66);
    k_expand<<<cdiv_i((long)16 * 4160, 256), 256, 0, stream>>>(
        dec1_w, dec1_b, Wdec1, Bdec1, 2, 25, 5, 13, 1, 16, 64, 65);

    { // enc0: X0 -> X1
      ConvGeo g{X0, Wenc0, Benc0, X1, NB, 32, 3640, 16, 28, 3616, 64, 64,
                13, 2, 0, 0, 65, 66, 1056, 1};
      dim3 grid(cdiv_i((g.Tout + 15) >> 4, 8), g.Fout, g.B);
      k_conv_wmma<4><<<grid, 256, 4 * 16 * KSUP * 2, stream>>>(g);
    }
    { // enc1: X1 -> X2
      ConvGeo g{X1, Wenc1, Benc1, X2, NB, 28, 3616, 64, 24, 3592, 112, 112,
                13, 2, 0, 0, 65, 65, 4160, 1};
      dim3 grid(cdiv_i((g.Tout + 15) >> 4, 8), g.Fout, g.B);
      k_conv_wmma<7><<<grid, 256, 7 * 16 * KSUP * 2, stream>>>(g);
    }
    { // dec0 (convT): X2 -> X1
      ConvGeo g{X2, Wdec0, Bdec0, X1, NB, 24, 3592, 112, 28, 3616, 64, 64,
                13, 2, -4, -24, 65, 66, 7392, 1};
      dim3 grid(cdiv_i((g.Tout + 15) >> 4, 8), g.Fout, g.B);
      k_conv_wmma<4><<<grid, 256, 4 * 16 * KSUP * 2, stream>>>(g);
    }
    { // dec1 (convT): X1 -> X4
      ConvGeo g{X1, Wdec1, Bdec1, X4, NB, 28, 3616, 64, 32, 3640, 4, 4,
                13, 2, -4, -24, 65, 65, 4160, 1};
      dim3 grid(cdiv_i((g.Tout + 15) >> 4, 8), g.Fout, g.B);
      k_conv_wmma<1><<<grid, 256, 1 * 16 * KSUP * 2, stream>>>(g);
    }
    { // fin + crop + mix
      long tot = (long)NB * NCH * NF * NS * NT;
      k_fin<<<cdiv_i(tot, 256), 256, 0, stream>>>(X4, fin_w, fin_b, x, (float*)d_out, br);
    }
  }
}
